// FarFieldHead_10170482557400
// MI455X (gfx1250) — matrix operand order repaired and checked by reading, not compiled, not run on hardware
//
#include <hip/hip_runtime.h>
#include <cmath>

typedef __attribute__((ext_vector_type(16))) _Float16 v16h;
typedef __attribute__((ext_vector_type(8)))  _Float16 v8h;
typedef __attribute__((ext_vector_type(8)))  float    v8f;

#define N_LED   2048
#define RSS_DIM 12
#define HEAD    64
#define TILE    256        // LEDs per tile, one per thread
#define NTILES  (N_LED / TILE)
#define KSTRIDE 72         // f16 per Ktile row: [0..31]=X input (phase A) then [0..63]=K data (layer-2),
                           // bias f32 @ half-offset 64; 144B rows, 16B-aligned
#define HSTRIDE 40         // f16 per h row: 32 data + pad; 80B, 16B-aligned
#define SSTRIDE 2052       // f32 per score row (2048 + 4 pad)

// LDS layout (bytes); total 155840 -> 2 blocks per 320KB WGP
#define OFF_SCORES 0                          // 12*2052*4 = 98496
#define OFF_KTILE  98496                      // 256*72*2  = 36864
#define OFF_HTILE  (OFF_KTILE + 36864)        // 256*40*2  = 20480
#define SMEM_TOTAL (OFF_HTILE + 20480)        // 155840

// prologue-only staging, overlaid on Htile (all consumed into registers before tile loop)
#define OFF_P_QH   (OFF_HTILE + 0)            // 16*32*4 = 2048
#define OFF_P_XW   (OFF_HTILE + 2048)         // 16*4    = 64
#define OFF_P_QF16 (OFF_HTILE + 2112)         // 16*64*2 = 2048
#define OFF_P_KW2H (OFF_HTILE + 4160)         // 64*32*2 = 4096
#define OFF_P_KW1H (OFF_HTILE + 8256)         // 32*32*2 = 2048 (k_w1 zero-padded K=11->32)
#define OFF_P_KB1  (OFF_HTILE + 10304)        // 32*4    = 128
#define OFF_P_KB2  (OFF_HTILE + 10432)        // 64*4    = 256

static __device__ __forceinline__ v16h cat8(v8h lo, v8h hi) {
    return __builtin_shufflevector(lo, hi, 0,1,2,3,4,5,6,7,8,9,10,11,12,13,14,15);
}

__global__ void __launch_bounds__(256)
farfield_attn_kernel(const float* __restrict__ rss_t,
                     const float* __restrict__ led_features,
                     const float* __restrict__ led_positions,
                     const float* __restrict__ prev_pos,
                     const int*   __restrict__ freq_mask,
                     const float* __restrict__ q_w1, const float* __restrict__ q_b1,
                     const float* __restrict__ q_w2, const float* __restrict__ q_b2,
                     const float* __restrict__ k_w1, const float* __restrict__ k_b1,
                     const float* __restrict__ k_w2, const float* __restrict__ k_b2,
                     const float* __restrict__ sigma,
                     float* __restrict__ out_attn,
                     float* __restrict__ out_ww)
{
    extern __shared__ char smem_raw[];
    float*    s_scores = (float*)   (smem_raw + OFF_SCORES);   // [12][SSTRIDE]
    _Float16* s_Ktile  = (_Float16*)(smem_raw + OFF_KTILE);    // [256][KSTRIDE]
    _Float16* s_Htile  = (_Float16*)(smem_raw + OFF_HTILE);    // [256][HSTRIDE]
    // prologue staging (overlaid on Htile)
    float*    s_qh     = (float*)   (smem_raw + OFF_P_QH);
    float*    s_xw     = (float*)   (smem_raw + OFF_P_XW);
    _Float16* s_Qf16   = (_Float16*)(smem_raw + OFF_P_QF16);
    _Float16* s_kw2h   = (_Float16*)(smem_raw + OFF_P_KW2H);
    _Float16* s_kw1h   = (_Float16*)(smem_raw + OFF_P_KW1H);
    float*    s_kb1    = (float*)   (smem_raw + OFF_P_KB1);
    float*    s_kb2    = (float*)   (smem_raw + OFF_P_KB2);

    const int t    = threadIdx.x;
    const int b    = blockIdx.x;
    const int lane = t & 31;
    const int wave = t >> 5;
    const int half = lane >> 4;
    const int nn   = lane & 15;

    const float sg = sigma[0];
    const float inv2s2 = 0.5f / (sg * sg);

    // ---- stage weights into LDS as f16 (k_w1 K-padded 11->32 with zeros) ----
    for (int i = t; i < 32 * 32; i += 256) {
        int r = i >> 5, c = i & 31;
        s_kw1h[i] = (c < 11) ? (_Float16)k_w1[r * 11 + c] : (_Float16)0.0f;
    }
    for (int i = t; i < 64 * 32; i += 256) s_kw2h[i] = (_Float16)k_w2[i];
    if (t < 32) s_kb1[t] = k_b1[t];
    if (t < 64) s_kb2[t] = k_b2[t];

    // ---- weak-signal gating + x = rss*ww (rows 12..15 zero pad) ----
    if (t < 16) {
        float x = 0.0f;
        if (t < RSS_DIM) {
            float r  = rss_t[(size_t)b * RSS_DIM + t];
            float ww = __expf(-fabsf(r - 2.75f) * 0.5f) * (r > 0.5f ? 1.0f : 0.0f);
            out_ww[(size_t)b * RSS_DIM + t] = ww;
            x = r * ww;
        }
        s_xw[t] = x;
    }
    __syncthreads();

    // ---- q hidden: tanh(x*w1 + b1), [16][32] ----
    for (int idx = t; idx < 16 * 32; idx += 256) {
        int m = idx >> 5, j = idx & 31;
        s_qh[idx] = tanhf(s_xw[m] * q_w1[j] + q_b1[j]);
    }
    __syncthreads();

    // ---- Q = qh @ q_w2^T + b2 -> f16 [16][64] ----
    for (int idx = t; idx < 16 * HEAD; idx += 256) {
        int m = idx >> 6, o = idx & 63;
        float acc = q_b2[o];
        const float* w = &q_w2[o * 32];
        const float* h = &s_qh[m * 32];
        #pragma unroll
        for (int j = 0; j < 32; ++j) acc += w[j] * h[j];
        s_Qf16[idx] = (_Float16)acc;
    }
    __syncthreads();

    // ---- hoist all constant WMMA fragments into registers ----
    const _Float16* qrow = s_Qf16 + nn * HEAD;
    v16h QA1 = cat8(*(const v8h*)(qrow +      8 * half),
                    *(const v8h*)(qrow + 16 + 8 * half));     // head 0..31
    v16h QA2 = cat8(*(const v8h*)(qrow + 32 + 8 * half),
                    *(const v8h*)(qrow + 48 + 8 * half));     // head 32..63
    v16h WA[4];                                               // k_w2 head-chunks (M=16, K=32)
    #pragma unroll
    for (int ch = 0; ch < 4; ++ch) {
        const _Float16* wr = s_kw2h + (ch * 16 + nn) * 32;
        WA[ch] = cat8(*(const v8h*)(wr +      8 * half),
                      *(const v8h*)(wr + 16 + 8 * half));
    }
    v16h W1A[2];                                              // k_w1 out-chunks (M=16, K=32 padded)
    #pragma unroll
    for (int ch = 0; ch < 2; ++ch) {
        const _Float16* wr = s_kw1h + (ch * 16 + nn) * 32;
        W1A[ch] = cat8(*(const v8h*)(wr +      8 * half),
                       *(const v8h*)(wr + 16 + 8 * half));
    }
    v8f KB[4];                                                // k_b2 slices (head = ch*16+8*half+r)
    #pragma unroll
    for (int ch = 0; ch < 4; ++ch) KB[ch] = *(const v8f*)(s_kb2 + ch * 16 + 8 * half);
    v8f K1B[2];                                               // k_b1 slices
    #pragma unroll
    for (int ch = 0; ch < 2; ++ch) K1B[ch] = *(const v8f*)(s_kb1 + ch * 16 + 8 * half);
    __syncthreads();   // staging area (overlaid on Htile) now dead

    const float ppx = prev_pos[b * 3 + 0];
    const float ppy = prev_pos[b * 3 + 1];
    const float ppz = prev_pos[b * 3 + 2];

    for (int tile = 0; tile < NTILES; ++tile) {
        // ---- phase A: load inputs, f16-pack X into Ktile[0..31], distance bias ----
        const int ledt = tile * TILE + t;
        const size_t fb = ((size_t)b * N_LED + ledt) * 8;
        const size_t pb = ((size_t)b * N_LED + ledt) * 3;
        if (tile + 1 < NTILES) {
            __builtin_prefetch(led_features + fb + (size_t)TILE * 8, 0, 1);
            __builtin_prefetch(led_positions + pb + (size_t)TILE * 3, 0, 1);
        }
        float in[11];
        #pragma unroll
        for (int i = 0; i < 8; ++i) in[i] = led_features[fb + i];
        float px = led_positions[pb + 0];
        float py = led_positions[pb + 1];
        float pz = led_positions[pb + 2];
        in[8] = px; in[9] = py; in[10] = pz;
        float dx = ppx - px, dy = ppy - py, dz = ppz - pz;

        _Float16* xrow = s_Ktile + (size_t)t * KSTRIDE;
        v8h x0, x1, zz = {};
        #pragma unroll
        for (int i = 0; i < 8; ++i) x0[i] = (_Float16)in[i];
        #pragma unroll
        for (int i = 0; i < 8; ++i) x1[i] = (i < 3) ? (_Float16)in[8 + i] : (_Float16)0.0f;
        *(v8h*)(xrow +  0) = x0;
        *(v8h*)(xrow +  8) = x1;
        *(v8h*)(xrow + 16) = zz;
        *(v8h*)(xrow + 24) = zz;
        *(float*)(xrow + 64) = -(dx*dx + dy*dy + dz*dz) * inv2s2;
        __syncthreads();

        // ---- per 16-LED subtile: layer1 WMMA -> Htile, layer2 WMMA -> Ktile, score WMMA ----
        // (one wave owns each subtile's rows; per-wave LDS ops are program-ordered)
        for (int s = wave; s < TILE / 16; s += 8) {
            const int rbase = s * 16 + nn;
            _Float16* kout = s_Ktile + rbase * KSTRIDE;
            _Float16* hrow = s_Htile + rbase * HSTRIDE;

            // layer-1: h = relu(X @ w1^T + b1), B = X^T (K=32 zero-padded)
            v16h BX = cat8(*(const v8h*)(kout + 8 * half),
                           *(const v8h*)(kout + 16 + 8 * half));
            #pragma unroll
            for (int ch = 0; ch < 2; ++ch) {
                v8f c = {};
                c = __builtin_amdgcn_wmma_f32_16x16x32_f16(false, W1A[ch], false, BX,
                                                           (short)0, c, false, false);
                v8h hh;
                #pragma unroll
                for (int r = 0; r < 8; ++r)
                    hh[r] = (_Float16)fmaxf(c[r] + K1B[ch][r], 0.0f);
                *(v8h*)(hrow + ch * 16 + 8 * half) = hh;   // 8 consecutive h dims of LED row
            }

            // layer-2: K = h @ w2^T + b2, B = h^T (K=32)
            v16h BH = cat8(*(const v8h*)(hrow + 8 * half),
                           *(const v8h*)(hrow + 16 + 8 * half));
            #pragma unroll
            for (int ch = 0; ch < 4; ++ch) {
                v8f c = {};
                c = __builtin_amdgcn_wmma_f32_16x16x32_f16(false, WA[ch], false, BH,
                                                           (short)0, c, false, false);
                v8h kh;
                #pragma unroll
                for (int r = 0; r < 8; ++r) kh[r] = (_Float16)(c[r] + KB[ch][r]);
                *(v8h*)(kout + ch * 16 + 8 * half) = kh;   // overwrites X (already consumed)
            }

            // scores: S = Q . K^T over head=64 (two K-chunks)
            v16h B1 = cat8(*(const v8h*)(kout + 8 * half),
                           *(const v8h*)(kout + 16 + 8 * half));
            v16h B2 = cat8(*(const v8h*)(kout + 32 + 8 * half),
                           *(const v8h*)(kout + 48 + 8 * half));
            v8f cS = {};
            cS = __builtin_amdgcn_wmma_f32_16x16x32_f16(false, QA1, false, B1,
                                                        (short)0, cS, false, false);
            cS = __builtin_amdgcn_wmma_f32_16x16x32_f16(false, QA2, false, B2,
                                                        (short)0, cS, false, false);
            const int led  = tile * TILE + s * 16 + nn;
            const float bv = *(const float*)(kout + 64);
            #pragma unroll
            for (int r = 0; r < 8; ++r) {
                int M = r + 8 * half;                 // C-layout: M = r + 8*half
                if (M < RSS_DIM) {
                    int mv = freq_mask[M * N_LED + led];
                    s_scores[M * SSTRIDE + led] = (mv == 0) ? -3.0e38f : (cS[r] + bv);
                }
            }
        }
        __syncthreads();
    }

    // ---- softmax over n_led per row; one wave owns a row ----
    for (int m = wave; m < RSS_DIM; m += 8) {
        float* row = s_scores + m * SSTRIDE;
        float mx = -3.4e38f;
        for (int j = lane; j < N_LED; j += 32) mx = fmaxf(mx, row[j]);
        #pragma unroll
        for (int o = 16; o > 0; o >>= 1) mx = fmaxf(mx, __shfl_xor(mx, o, 32));
        float sum = 0.0f;
        for (int j = lane; j < N_LED; j += 32) {
            float e = __expf(row[j] - mx);
            row[j] = e;
            sum += e;
        }
        #pragma unroll
        for (int o = 16; o > 0; o >>= 1) sum += __shfl_xor(sum, o, 32);
        const float inv = 1.0f / sum;
        float* orow = out_attn + ((size_t)b * RSS_DIM + m) * N_LED;
        for (int j = lane; j < N_LED; j += 32) orow[j] = row[j] * inv;
    }
}

extern "C" void kernel_launch(void* const* d_in, const int* in_sizes, int n_in,
                              void* d_out, int out_size, void* d_ws, size_t ws_size,
                              hipStream_t stream) {
    const float* rss_t         = (const float*)d_in[0];
    const float* led_features  = (const float*)d_in[1];
    const float* led_positions = (const float*)d_in[2];
    const float* prev_pos      = (const float*)d_in[3];
    const int*   freq_mask     = (const int*)  d_in[4];
    const float* q_w1          = (const float*)d_in[5];
    const float* q_b1          = (const float*)d_in[6];
    const float* q_w2          = (const float*)d_in[7];
    const float* q_b2          = (const float*)d_in[8];
    const float* k_w1          = (const float*)d_in[9];
    const float* k_b1          = (const float*)d_in[10];
    const float* k_w2          = (const float*)d_in[11];
    const float* k_b2          = (const float*)d_in[12];
    const float* sigma         = (const float*)d_in[13];

    const int B = in_sizes[0] / RSS_DIM;            // 2048
    float* out_attn = (float*)d_out;
    float* out_ww   = out_attn + (size_t)B * RSS_DIM * N_LED;

    farfield_attn_kernel<<<dim3(B), dim3(256), SMEM_TOTAL, stream>>>(
        rss_t, led_features, led_positions, prev_pos, freq_mask,
        q_w1, q_b1, q_w2, q_b2, k_w1, k_b1, k_w2, k_b2, sigma,
        out_attn, out_ww);
}
